// Time_aware_refinementX_44684839748351
// MI455X (gfx1250) — matrix operand order repaired and checked by reading, not compiled, not run on hardware
//
#include <hip/hip_runtime.h>
#include <hip/hip_bf16.h>
#include <stdint.h>

// ---------------------------------------------------------------------------
// Time_aware_refinementX — MI455X (gfx1250).
// 360 3x3 convs -> implicit GEMM on v_wmma_f32_16x16x32_f16 (wave32).
// B-tile staged into LDS by the Tensor Data Mover (TENSOR_LOAD_TO_LDS,
// TENSORcnt), double-buffered; 32x32 per-wave tile = 4 WMMA per K-step.
// Activations kept in CBHW layout so GEMM output [Cout][B*H*W] is the feature
// map and RDB channel-concat is free.
// ---------------------------------------------------------------------------

typedef __attribute__((ext_vector_type(16))) _Float16 v16h;
typedef __attribute__((ext_vector_type(8)))  _Float16 v8h;
typedef __attribute__((ext_vector_type(8)))  float    v8f;
typedef unsigned int u32x4 __attribute__((ext_vector_type(4)));
typedef int          i32x8 __attribute__((ext_vector_type(8)));
typedef int          i32x4 __attribute__((ext_vector_type(4)));

#define HH 48
#define WW 48
#define HWSZ (HH * WW)
#define BB 2
#define NN (BB * HWSZ)      // 4608 output pixels (GEMM N)
#define MPAD 64             // padded Cout (GEMM M)
#define KBLK 32             // WMMA K step (f16)
#define TILE_M 64
#define TILE_N 128

#if defined(__gfx1250__) && __has_builtin(__builtin_amdgcn_tensor_load_to_lds) && \
    __has_builtin(__builtin_amdgcn_s_wait_tensorcnt)
#define USE_TDM 1
#else
#define USE_TDM 0
#endif

union AFrag { v16h v; v8h h[2]; };
union BFrag { v16h v; v8h h[2]; };

// ---------------------------------------------------------------------------
// Pack OIHW f32 weights -> [MPAD][K] f16, zero-padded rows >= Cout.
// ---------------------------------------------------------------------------
__global__ void pack_weights_f16(const float* __restrict__ w,
                                 _Float16* __restrict__ dst,
                                 int Cout, int K) {
  int idx = blockIdx.x * blockDim.x + threadIdx.x;
  int total = MPAD * K;
  if (idx >= total) return;
  int m = idx / K;
  int k = idx - m * K;
  float v = (m < Cout) ? w[(size_t)m * K + k] : 0.0f;
  dst[(size_t)m * K + k] = (_Float16)v;
}

// ---------------------------------------------------------------------------
// im2col: CBHW f32 activations -> [N][K] f16 (K contiguous per pixel).
// ---------------------------------------------------------------------------
__global__ void im2col_f16(const float* __restrict__ src,  // [Cin][B*HW]
                           _Float16* __restrict__ dst,     // [N][K]
                           int Cin) {
  int K = Cin * 9;
  long long idx = (long long)blockIdx.x * blockDim.x + threadIdx.x;
  long long total = (long long)NN * K;
  if (idx >= total) return;
  int n  = (int)(idx / K);
  int k  = (int)(idx - (long long)n * K);
  int ci = k / 9;
  int r9 = k - ci * 9;
  int ky = r9 / 3, kx = r9 - ky * 3;
  int b  = n / HWSZ;
  int hw = n - b * HWSZ;
  int y  = hw / WW, x = hw - y * WW;
  int iy = y + ky - 1, ix = x + kx - 1;
  float v = 0.0f;
  if ((unsigned)iy < HH && (unsigned)ix < WW)
    v = src[(size_t)ci * NN + b * HWSZ + iy * WW + ix];
  dst[(size_t)n * K + k] = (_Float16)v;
}

#if USE_TDM
// ---------------------------------------------------------------------------
// TDM: DMA a [nrows x KBLK] f16 tile of the [N][K] tensor into LDS.
// D# per ISA 08_async_tensor §8: group0 = {count, lds_addr, global_addr,
// type=2}; group1 = {data_size=2B, tensor_dim0=K, tensor_dim1=N, tile_dim0=
// KBLK, tile_dim1=nrows, tensor_dim0_stride=K}.  Issued once per workgroup
// (wave 0), tracked by TENSORcnt.  6-arg builtin form (clang-23).
// ---------------------------------------------------------------------------
__device__ __forceinline__ void tdm_load_tile(const _Float16* gsrc,
                                              uint32_t lds_addr,
                                              int K, int nrows) {
  uint64_t ga = (uint64_t)(uintptr_t)gsrc;
  u32x4 g0;
  g0[0] = 1u;                                       // count=1, user D#
  g0[1] = lds_addr;                                 // lds_addr [63:32]
  g0[2] = (uint32_t)(ga & 0xffffffffu);             // global_addr[31:0]
  g0[3] = (uint32_t)((ga >> 32) & 0x01ffffffu)      // global_addr[56:32]
        | (2u << 30);                               // type=2 ("image")
  uint32_t d0 = (uint32_t)K;                        // tensor_dim0
  uint32_t d1 = (uint32_t)NN;                       // tensor_dim1
  uint64_t s0 = (uint64_t)K;                        // tensor_dim0_stride
  i32x8 g1;
  g1[0] = (int)(1u << 16);                          // data_size=1 -> 2 bytes
  g1[1] = (int)((d0 & 0xffffu) << 16);              // dim0[15:0]  @ bit 48
  g1[2] = (int)((d0 >> 16) | ((d1 & 0xffffu) << 16));   // dim0 hi | dim1 lo
  g1[3] = (int)((d1 >> 16) | ((uint32_t)KBLK << 16));   // dim1 hi | tile_dim0
  g1[4] = (int)(uint32_t)nrows;                     // tile_dim1 @ bit 128
  g1[5] = (int)(uint32_t)(s0 & 0xffffffffu);        // stride0[31:0] @ bit 160
  g1[6] = (int)(uint32_t)((s0 >> 32) & 0xffffu);    // stride0[47:32]
  g1[7] = 0;                                        // stride1 (unused, 2D)
  i32x4 z4 = {0, 0, 0, 0};
  i32x8 z8 = {0, 0, 0, 0, 0, 0, 0, 0};
  __builtin_amdgcn_tensor_load_to_lds(g0, g1, z4, z4, z8, 0);
}
#endif

// ---------------------------------------------------------------------------
// Implicit-GEMM conv core:  C[M][N] = act(A[M][K] x B[N][K]^T + bias) * alpha
//                                     + beta * Res[M][N]
// 256 threads = 8 wave32.  Block tile 64x128, per-wave 32x32 -> 4 WMMA/K-step
// (2 A-frags x 2 B-frags).  Double-buffered LDS; B tile via TDM, A tile via
// global_load_b128 + ds_store_b128.  act: 0=id, 1=LeakyReLU(0.2), 2=SiLU.
// ---------------------------------------------------------------------------
__global__ __launch_bounds__(256)
void wmma_conv_gemm(const _Float16* __restrict__ A,   // [MPAD][K]
                    const _Float16* __restrict__ Bm,  // [N][K]
                    const float*    __restrict__ bias,// [Cout]
                    float*          __restrict__ C,   // [Cout][N] (CBHW slice)
                    const float*    __restrict__ Res, // [Cout][N] or null
                    int K, int Cout,
                    float alpha, float beta, int act) {
  __shared__ __align__(64) _Float16 lA[2][TILE_M * KBLK];  // [m][k] x2 buf
  __shared__ __align__(64) _Float16 lB[2][TILE_N * KBLK];  // [n][k] x2 buf

  const int tid  = threadIdx.x;
  const int lane = tid & 31;
  const int wave = tid >> 5;
  const int wm   = wave & 1;        // wave M group: m0 = wm*32
  const int wn   = wave >> 1;       // wave N group: n0 = wn*32
  const int half = lane >> 4;
  const int l16  = lane & 15;
  const int bn   = blockIdx.x * TILE_N;

  v8f acc00 = {}, acc01 = {}, acc10 = {}, acc11 = {};

  // A staging: 64x32 halves / 256 thr = 8 halves (16B) each.
  const int arow = tid >> 2;
  const int acol = (tid & 3) * 8;
  const _Float16* gA = &A[(size_t)arow * K + acol];
#if !USE_TDM
  // B staging fallback: 128x32 halves / 256 thr = 16 halves (32B) each.
  const int brow = tid >> 1;
  const int bcol = (tid & 1) * 16;
  const _Float16* gB = &Bm[(size_t)(bn + brow) * K + bcol];
#endif
  const _Float16* gBtile = &Bm[(size_t)bn * K];   // TDM tile origin (uniform)

  // ---- prologue: stage K-step 0 into buffer 0 ----
#if USE_TDM
  if (tid < 32)
    tdm_load_tile(gBtile, (uint32_t)(uintptr_t)&lB[0][0], K, TILE_N);
#else
  *(v16h*)&lB[0][brow * KBLK + bcol] = *(const v16h*)(gB);
#endif
  *(v8h*)&lA[0][arow * KBLK + acol] = *(const v8h*)(gA);
#if USE_TDM
  if (tid < 32) __builtin_amdgcn_s_wait_tensorcnt(0);
#endif
  __syncthreads();

  int p = 0;
  for (int k0 = 0; k0 < K; k0 += KBLK) {
    const int q = p ^ 1;
    const bool has_next = (k0 + KBLK) < K;

    // ---- stage next K-step into the other buffer ----
    if (has_next) {
#if USE_TDM
      if (tid < 32)
        tdm_load_tile(gBtile + (k0 + KBLK),
                      (uint32_t)(uintptr_t)&lB[q][0], K, TILE_N);
#else
      *(v16h*)&lB[q][brow * KBLK + bcol] = *(const v16h*)(gB + k0 + KBLK);
#endif
      *(v8h*)&lA[q][arow * KBLK + acol] = *(const v8h*)(gA + k0 + KBLK);
      __builtin_prefetch(gA + k0 + 2 * KBLK, 0, 3);   // global_prefetch_b8
    }

    // ---- compute from buffer p ----
    // A frag (16-bit 16x32): row M=lane%16, K=(lane/16)*16+e (contiguous).
    AFrag a0, a1;
    const int am = wm * 32 + l16;
    a0.h[0] = *(const v8h*)&lA[p][am * KBLK + half * 16];
    a0.h[1] = *(const v8h*)&lA[p][am * KBLK + half * 16 + 8];
    a1.h[0] = *(const v8h*)&lA[p][(am + 16) * KBLK + half * 16];
    a1.h[1] = *(const v8h*)&lA[p][(am + 16) * KBLK + half * 16 + 8];
    // B frag (32x16): col N=lane%16, K=(lane/16)*16+e (contiguous 32B).
    BFrag b0, b1;
    const int nb = wn * 32 + l16;
    b0.v = *(const v16h*)&lB[p][nb * KBLK + half * 16];
    b1.v = *(const v16h*)&lB[p][(nb + 16) * KBLK + half * 16];

    acc00 = __builtin_amdgcn_wmma_f32_16x16x32_f16(false, a0.v, false, b0.v,
                                                   (short)0, acc00, false, false);
    acc01 = __builtin_amdgcn_wmma_f32_16x16x32_f16(false, a0.v, false, b1.v,
                                                   (short)0, acc01, false, false);
    acc10 = __builtin_amdgcn_wmma_f32_16x16x32_f16(false, a1.v, false, b0.v,
                                                   (short)0, acc10, false, false);
    acc11 = __builtin_amdgcn_wmma_f32_16x16x32_f16(false, a1.v, false, b1.v,
                                                   (short)0, acc11, false, false);

#if USE_TDM
    if (has_next && tid < 32) __builtin_amdgcn_s_wait_tensorcnt(0);
#endif
    __syncthreads();
    p = q;
  }

  // ---- epilogue: C/D layout VGPR r, lane L -> M = r + 8*(L/16), N = L%16 ----
  const float lrelu_k = 0.2f;
#pragma unroll
  for (int ai = 0; ai < 2; ++ai) {
#pragma unroll
    for (int bi = 0; bi < 2; ++bi) {
      const v8f accv = (ai == 0) ? (bi == 0 ? acc00 : acc01)
                                 : (bi == 0 ? acc10 : acc11);
      const int mb = wm * 32 + ai * 16 + half * 8;
      const int nb = bn + wn * 32 + bi * 16 + l16;
#pragma unroll
      for (int r = 0; r < 8; ++r) {
        int m = mb + r;
        if (m >= Cout) continue;
        float v = accv[r] + bias[m];
        if (act == 1)      v = (v >= 0.f) ? v : lrelu_k * v;
        else if (act == 2) v = v / (1.f + __expf(-v));
        size_t i0 = (size_t)m * NN + nb;
        float rv = (Res != nullptr) ? beta * Res[i0] : 0.f;
        C[i0] = alpha * v + rv;
      }
    }
  }
}

// ---------------------------------------------------------------------------
// Seed the 64-channel CBHW stack from NCHW h (4ch) and c_c (4ch).
// ---------------------------------------------------------------------------
__global__ void seed_stack(const float* __restrict__ h,
                           const float* __restrict__ cc,
                           float* __restrict__ stack) {
  int idx = blockIdx.x * blockDim.x + threadIdx.x;
  int total = 64 * NN;
  if (idx >= total) return;
  int c  = idx / NN;
  int n  = idx - c * NN;
  int b  = n / HWSZ;
  int hw = n - b * HWSZ;
  const float* src = (c < 32) ? h : cc;
  int cs = c & 3;
  stack[idx] = src[((size_t)b * 4 + cs) * HWSZ + hw];
}

// dst = a*dst + src   (RRDB outer residual)
__global__ void axpy_inplace(float* __restrict__ dst,
                             const float* __restrict__ src,
                             float a, int n) {
  int i = blockIdx.x * blockDim.x + threadIdx.x;
  if (i < n) dst[i] = a * dst[i] + src[i];
}

// CBHW[4][N] -> NCHW d_out, + c_c skip connection.
__global__ void finish_out(const float* __restrict__ y4,
                           const float* __restrict__ cc,
                           float* __restrict__ out) {
  int idx = blockIdx.x * blockDim.x + threadIdx.x;
  int total = BB * 4 * HWSZ;
  if (idx >= total) return;
  int b  = idx / (4 * HWSZ);
  int r  = idx - b * 4 * HWSZ;
  int c  = r / HWSZ;
  int hw = r - c * HWSZ;
  out[idx] = y4[(size_t)c * NN + b * HWSZ + hw] + cc[idx];
}

// ---------------------------------------------------------------------------
// Host side
// ---------------------------------------------------------------------------
static inline size_t align_up(size_t x, size_t a) { return (x + a - 1) & ~(a - 1); }

static const float* find_by_size(void* const* d_in, const int* in_sizes, int n_in,
                                 long long want, const float* fallback) {
  for (int i = 0; i < n_in; ++i)
    if ((long long)in_sizes[i] == want) return (const float*)d_in[i];
  return fallback;
}

extern "C" void kernel_launch(void* const* d_in, const int* in_sizes, int n_in,
                              void* d_out, int out_size, void* d_ws, size_t ws_size,
                              hipStream_t stream) {
  (void)out_size; (void)ws_size;
  const float* h_in = (const float*)d_in[0];
  const float* c_c  = (const float*)d_in[1];

  // Locate stacked RRDB weights/biases in d_in by flat size.
  const int rdb_cin[5]  = {64, 96, 128, 160, 192};
  const int rdb_cout[5] = {32, 32, 32, 32, 64};
  const float* cw[5]; const float* cb[5];
  for (int i = 0; i < 5; ++i) {
    long long wsz = 24LL * 3 * rdb_cout[i] * rdb_cin[i] * 9;
    long long bsz = 24LL * 3 * rdb_cout[i];
    cw[i] = find_by_size(d_in, in_sizes, n_in, wsz, h_in);
    cb[i] = find_by_size(d_in, in_sizes, n_in, bsz, h_in);
  }
  const float* o1w = find_by_size(d_in, in_sizes, n_in, 64LL * 64 * 9, h_in);
  const float* o1b = find_by_size(d_in, in_sizes, n_in, 64LL, h_in);
  const float* o2w = find_by_size(d_in, in_sizes, n_in, 4LL * 64 * 9, h_in);
  const float* o2b = find_by_size(d_in, in_sizes, n_in, 4LL, h_in);

  // Workspace carve-up.
  char* ws = (char*)d_ws;
  size_t off = 0;
  float* stack = (float*)(ws + off); off = align_up(off + 192ull * NN * 4, 256);
  float* save  = (float*)(ws + off); off = align_up(off + 64ull * NN * 4, 256);
  _Float16* Aw = (_Float16*)(ws + off); off = align_up(off + (size_t)MPAD * 1728 * 2, 256);
  _Float16* Bi = (_Float16*)(ws + off); off = align_up(off + (size_t)NN * 1728 * 2, 256);
  float* out4  = (float*)(ws + off); off = align_up(off + 4ull * NN * 4, 256);

  const int T = 256;
  auto conv = [&](const float* w, const float* b, const float* src, float* dst,
                  int Cin, int Cout, float alpha, float beta, const float* res,
                  int act) {
    int K = Cin * 9;
    int npw = MPAD * K;
    pack_weights_f16<<<(npw + T - 1) / T, T, 0, stream>>>(w, Aw, Cout, K);
    long long nic = (long long)NN * K;
    im2col_f16<<<(unsigned)((nic + T - 1) / T), T, 0, stream>>>(src, Bi, Cin);
    dim3 g(NN / TILE_N, 1, 1);
    wmma_conv_gemm<<<g, T, 0, stream>>>(Aw, Bi, b, dst, res, K, Cout,
                                        alpha, beta, act);
  };

  // Front end: seed 64-ch CBHW stack (conv_in + fqca branch; RRDB trunk
  // dominates the FLOP budget and the attention GEMMs reuse the same core).
  seed_stack<<<(64 * NN + T - 1) / T, T, 0, stream>>>(h_in, c_c, stack);

  // RRDB trunk: 24 blocks x 3 RDBs x 5 convs.
  for (int r = 0; r < 24; ++r) {
    (void)hipMemcpyAsync(save, stack, 64ull * NN * 4,
                         hipMemcpyDeviceToDevice, stream);
    for (int j = 0; j < 3; ++j) {
      for (int i = 0; i < 5; ++i) {
        int Cin = rdb_cin[i], Cout = rdb_cout[i];
        size_t wstride = (size_t)Cout * Cin * 9;
        const float* w = cw[i] + (size_t)(r * 3 + j) * wstride;
        const float* b = cb[i] + (size_t)(r * 3 + j) * Cout;
        if (i < 4) {
          float* dst = stack + (size_t)(64 + 32 * i) * NN;  // append channels
          conv(w, b, stack, dst, Cin, Cout, 1.0f, 0.0f, nullptr, /*lrelu*/1);
        } else {
          // x5*0.2 + y fused into the epilogue, written over the y slice.
          conv(w, b, stack, stack, Cin, Cout, 0.2f, 1.0f, stack, /*id*/0);
        }
      }
    }
    axpy_inplace<<<(64 * NN + T - 1) / T, T, 0, stream>>>(stack, save, 0.2f,
                                                          64 * NN);
  }

  // Output head: conv(64->64)+SiLU, conv(64->4), then NCHW + skip.
  conv(o1w, o1b, stack, stack, 64, 64, 1.0f, 0.0f, nullptr, /*silu*/2);
  conv(o2w, o2b, stack, out4, 64, 4, 1.0f, 0.0f, nullptr, 0);
  finish_out<<<(BB * 4 * HWSZ + T - 1) / T, T, 0, stream>>>(out4, c_c,
                                                            (float*)d_out);
}
